// STTLayer_48911087566882
// MI455X (gfx1250) — matrix operand order repaired and checked by reading, not compiled, not run on hardware
//
#include <hip/hip_runtime.h>
#include <hip/hip_bf16.h>

// ---------------------------------------------------------------------------
// STT layer for MI455X (gfx1250, wave32, WMMA). All GEMMs run through
// v_wmma_f32_16x16x32_f16 with fp32 accumulate. GEMM staging uses the CDNA5
// data movers: Tensor Data Mover (tensor_load_to_lds, TENSORcnt) for the
// activation tile and global_load_async_to_lds_b128 (ASYNCcnt) for the
// weight tile. Attention is a fused flash-style WMMA kernel.
// ---------------------------------------------------------------------------

typedef __attribute__((ext_vector_type(16))) _Float16 v16h;
typedef __attribute__((ext_vector_type(8)))  _Float16 v8h;
typedef __attribute__((ext_vector_type(4)))  _Float16 v4h;
typedef __attribute__((ext_vector_type(8)))  float    v8f;
typedef __attribute__((ext_vector_type(4)))  unsigned int v4u;
typedef __attribute__((ext_vector_type(8)))  int v8i;
typedef __attribute__((ext_vector_type(4)))  int v4i;

#define B_  4
#define T_C 1024
#define D_C 2048
#define H_C 16
#define HD_C 128
#define F_C 8192
#define KSEL 256
#define EPS_C 1e-6f

__device__ __forceinline__ v8f wmma_f16(v16h a, v16h b, v8f c) {
  return __builtin_amdgcn_wmma_f32_16x16x32_f16(false, a, false, b, (short)0, c,
                                                false, false);
}

__device__ __forceinline__ v16h load_frag_2x8(const float* p0, const float* p1,
                                              float s) {
  v16h r;
#pragma unroll
  for (int j = 0; j < 8; ++j) {
    r[j]     = (_Float16)(p0[j] * s);
    r[j + 8] = (_Float16)(p1[j] * s);
  }
  return r;
}

// ---------------------------------------------------------------------------
// Generic tiled GEMM: C[M,N] = A[M,K] @ W[K,N]  (+bias / +residual / swiglu)
// Block tile 128x128, K-step 32, 8 waves, each wave owns a 32x64 tile
// (2x4 of 16x16 WMMA accumulators).  mode: 0=plain(+bias), 1=+aux residual,
// 2 = silu(acc)*aux  (aux may alias C).
//
// A tile (128x32 fp32) arrives via TDM into As32 with LDS row stride 36
// floats (pad_interval = 32 dwords, pad_amount = 4 dwords).
// W tile (32x128 fp32) arrives via global_load_async_to_lds_b128 into Bst
// (row stride 132 floats), then a register-blocked transpose/convert pass
// produces the f16 [n][k] tile Bs.
// ---------------------------------------------------------------------------
__global__ __launch_bounds__(256, 1) void gemm_wmma_kernel(
    const float* __restrict__ A, const float* __restrict__ W,
    const float* __restrict__ bias, const float* __restrict__ aux,
    float* __restrict__ C, int M, int N, int Kd, int mode) {
  __shared__ __align__(16) float     As32[128 * 36];  // A tile fp32 (TDM)
  __shared__ __align__(16) float     Bst[32 * 132];   // W tile fp32 (async)
  __shared__ __align__(16) _Float16  Bs[128 * 40];    // W tile f16, [n][k]

  const int tid   = threadIdx.x;
  const int lane  = tid & 31;
  const int wave  = tid >> 5;
  const int waveM = wave >> 1;  // 0..3
  const int waveN = wave & 1;   // 0..1
  const int bM    = blockIdx.y * 128;
  const int bN    = blockIdx.x * 128;
  const int l16   = lane & 15;
  const int hi    = lane >> 4;  // 0/1

  v8f acc[2][4];
#pragma unroll
  for (int i = 0; i < 2; ++i)
#pragma unroll
    for (int j = 0; j < 4; ++j) acc[i][j] = (v8f){0, 0, 0, 0, 0, 0, 0, 0};

  const unsigned ldsA = (unsigned)(size_t)(&As32[0]);

  for (int k0 = 0; k0 < Kd; k0 += 32) {
    __syncthreads();  // previous iteration finished reading LDS

    // ---- A tile via Tensor Data Mover (wave 0 issues one descriptor) ----
    if (wave == 0) {
      unsigned long long ga =
          (unsigned long long)(size_t)(A + (size_t)bM * Kd + k0);
      // D# group 0: count=1, lds_addr, global_addr[56:0], type=2
      v4u g0 = {1u, ldsA, (unsigned)(ga & 0xFFFFFFFFu),
                (unsigned)((ga >> 32) & 0x01FFFFFFu) | (2u << 30)};
      // D# group 1: data_size=4B, pad_enable, pad_interval=32dw, pad_amount=4dw
      unsigned w0 = (2u << 16) | (1u << 20) | (4u << 22) | (3u << 25);
      unsigned w1 = ((unsigned)Kd & 0xFFFFu) << 16;           // tensor_dim0 lo
      unsigned w2 = (((unsigned)Kd >> 16) & 0xFFFFu) |        // tensor_dim0 hi
                    (((unsigned)M & 0xFFFFu) << 16);          // tensor_dim1 lo
      unsigned w3 = (((unsigned)M >> 16) & 0xFFFFu) |         // tensor_dim1 hi
                    (32u << 16);                              // tile_dim0 = 32
      unsigned w4 = 128u;                                     // tile_dim1 = 128
      unsigned w5 = (unsigned)Kd;                             // dim0_stride lo
      unsigned w6 = 0u;                                       // stride hi, etc
      unsigned w7 = 0u;
      v8i g1 = {(int)w0, (int)w1, (int)w2, (int)w3,
                (int)w4, (int)w5, (int)w6, (int)w7};
      v4i z4 = {0, 0, 0, 0};
      v8i z8 = {0, 0, 0, 0, 0, 0, 0, 0};
      __builtin_amdgcn_tensor_load_to_lds(g0, g1, z4, z4, z8, 0);
    }

    // ---- W tile via async global->LDS (raw fp32, no VGPR round trip) ----
#pragma unroll
    for (int i = 0; i < 4; ++i) {
      int f  = tid + i * 256;
      int kr = f >> 5;            // 0..31
      int c4 = (f & 31) << 2;     // 0..124
      unsigned lds_off = (unsigned)(size_t)(&Bst[kr * 132 + c4]);
      unsigned long long ga =
          (unsigned long long)(size_t)(W + (size_t)(k0 + kr) * N + bN + c4);
      asm volatile("global_load_async_to_lds_b128 %0, %1, off"
                   :: "v"(lds_off), "v"(ga)
                   : "memory");
    }
    asm volatile("s_wait_asynccnt 0" ::: "memory");
    __syncthreads();  // Bst visible to all waves

    // ---- transpose/convert W tile: Bst[k][n] fp32 -> Bs[n][k] f16 ----
    {
      int n0 = tid & 127;
      int kh = tid >> 7;  // 0/1 -> k = kh*16 + j
      __align__(16) _Float16 tmp[16];
#pragma unroll
      for (int j = 0; j < 16; ++j)
        tmp[j] = (_Float16)Bst[(kh * 16 + j) * 132 + n0];
      v8h* dst = reinterpret_cast<v8h*>(&Bs[n0 * 40 + kh * 16]);
      dst[0] = *reinterpret_cast<v8h*>(&tmp[0]);
      dst[1] = *reinterpret_cast<v8h*>(&tmp[8]);
    }
    if (wave == 0) __builtin_amdgcn_s_wait_tensorcnt(0);
    __syncthreads();  // As32 + Bs ready

    // A fragments (fp32 LDS -> f16): lane l16 = row, kbase = 16*hi;
    // elems 0..15 = K kb..kb+15
    v16h af[2];
#pragma unroll
    for (int mt = 0; mt < 2; ++mt) {
      int arow = waveM * 32 + mt * 16 + l16;
      int kb   = hi ? 16 : 0;
      const float* ap = &As32[arow * 36 + kb];
      af[mt] = load_frag_2x8(ap, ap + 8, 1.0f);
    }
    // B fragments: col = l16, kbase = 16*hi; 16 contiguous K values
    v16h bf[4];
#pragma unroll
    for (int nt = 0; nt < 4; ++nt) {
      int bcol = waveN * 64 + nt * 16 + l16;
      int kb2  = hi ? 16 : 0;
      v8h* d = reinterpret_cast<v8h*>(&bf[nt]);
      d[0] = *reinterpret_cast<const v8h*>(&Bs[bcol * 40 + kb2]);
      d[1] = *reinterpret_cast<const v8h*>(&Bs[bcol * 40 + kb2 + 8]);
    }
#pragma unroll
    for (int mt = 0; mt < 2; ++mt)
#pragma unroll
      for (int nt = 0; nt < 4; ++nt)
        acc[mt][nt] = wmma_f16(af[mt], bf[nt], acc[mt][nt]);
  }

  // epilogue (C 16x16 layout: vgpr r -> row r (lanes 0-15) / r+8 (16-31))
#pragma unroll
  for (int mt = 0; mt < 2; ++mt) {
#pragma unroll
    for (int nt = 0; nt < 4; ++nt) {
      int gn   = bN + waveN * 64 + nt * 16 + l16;
      float bv = bias ? bias[gn] : 0.0f;
#pragma unroll
      for (int r = 0; r < 8; ++r) {
        int gm     = bM + waveM * 32 + mt * 16 + r + hi * 8;
        size_t idx = (size_t)gm * N + gn;
        float v    = acc[mt][nt][r] + bv;
        if (mode == 1)
          v += aux[idx];
        else if (mode == 2)
          v = (v / (1.0f + __expf(-v))) * aux[idx];  // silu(g)*u
        C[idx] = v;
      }
    }
  }
}

// ---------------------------------------------------------------------------
// Flash attention (causal). Layout [B,T,H,HD] fp32. Each wave owns a 16-query
// tile; 32-key blocks; S=QK^T and O+=PV via WMMA; online softmax with 16-lane
// shuffle reductions; P reshaped through per-wave LDS.
// ---------------------------------------------------------------------------
__global__ __launch_bounds__(128, 1) void flash_attn_kernel(
    const float* __restrict__ Q, const float* __restrict__ Km,
    const float* __restrict__ Vm, float* __restrict__ O, int T, int H) {
  const int HD   = HD_C;
  const int lane = threadIdx.x & 31;
  const int wave = threadIdx.x >> 5;
  const int l16  = lane & 15;
  const int hi   = lane >> 4;
  const int bh   = blockIdx.y;
  const int b    = bh / H;
  const int h    = bh % H;
  const int qbase = blockIdx.x * 64 + wave * 16;
  __shared__ __align__(16) _Float16 Plds[4][16][32];

  const float scale = 0.08838834764831845f;  // 1/sqrt(128), folded into Q

  v16h qf[4];
  {
    int qr = qbase + l16;
    const float* qp = Q + ((size_t)(b * T + qr) * H + h) * HD;
    int kb = hi ? 16 : 0;
#pragma unroll
    for (int c = 0; c < 4; ++c)
      qf[c] = load_frag_2x8(qp + c * 32 + kb, qp + c * 32 + kb + 8, scale);
  }

  float mrow[8], lrow[8];
  v8f o_acc[8];
#pragma unroll
  for (int r = 0; r < 8; ++r) { mrow[r] = -1e30f; lrow[r] = 0.0f; }
#pragma unroll
  for (int nt = 0; nt < 8; ++nt) o_acc[nt] = (v8f){0, 0, 0, 0, 0, 0, 0, 0};

  const int nblk = ((qbase + 15) >> 5) + 1;  // causal: keys <= qbase+15
  for (int blk = 0; blk < nblk; ++blk) {
    int kp0 = blk * 32;
    v8f s0 = (v8f){0, 0, 0, 0, 0, 0, 0, 0};
    v8f s1 = (v8f){0, 0, 0, 0, 0, 0, 0, 0};
    int kb2 = hi ? 16 : 0;
#pragma unroll
    for (int c = 0; c < 4; ++c) {
      const float* kp0p =
          Km + ((size_t)(b * T + kp0 + l16) * H + h) * HD + c * 32 + kb2;
      v16h kf0 = load_frag_2x8(kp0p, kp0p + 8, 1.0f);
      s0 = wmma_f16(qf[c], kf0, s0);
      const float* kp1p =
          Km + ((size_t)(b * T + kp0 + 16 + l16) * H + h) * HD + c * 32 + kb2;
      v16h kf1 = load_frag_2x8(kp1p, kp1p + 8, 1.0f);
      s1 = wmma_f16(qf[c], kf1, s1);
    }
    // masked online softmax over this 32-key block
#pragma unroll
    for (int r = 0; r < 8; ++r) {
      int qpos = qbase + r + hi * 8;
      float a = s0[r];
      if (kp0 + l16 > qpos) a = -1e30f;
      float c2 = s1[r];
      if (kp0 + 16 + l16 > qpos) c2 = -1e30f;
      float rm = fmaxf(a, c2);
#pragma unroll
      for (int m = 1; m < 16; m <<= 1) rm = fmaxf(rm, __shfl_xor(rm, m));
      float mnew = fmaxf(mrow[r], rm);
      float corr = __expf(mrow[r] - mnew);
      float p0 = __expf(a - mnew);
      float p1 = __expf(c2 - mnew);
      float ps = p0 + p1;
#pragma unroll
      for (int m = 1; m < 16; m <<= 1) ps += __shfl_xor(ps, m);
      lrow[r] = lrow[r] * corr + ps;
      mrow[r] = mnew;
#pragma unroll
      for (int nt = 0; nt < 8; ++nt) o_acc[nt][r] *= corr;
      int rl = r + hi * 8;
      Plds[wave][rl][l16]      = (_Float16)p0;
      Plds[wave][rl][16 + l16] = (_Float16)p1;
    }
    __builtin_amdgcn_wave_barrier();
    asm volatile("s_wait_dscnt 0" ::: "memory");
    __builtin_amdgcn_wave_barrier();
    // P as a 16x32 A-fragment
    v16h pf;
    {
      int kb = hi ? 16 : 0;
      v8h* d = reinterpret_cast<v8h*>(&pf);
      d[0] = *reinterpret_cast<const v8h*>(&Plds[wave][l16][kb]);
      d[1] = *reinterpret_cast<const v8h*>(&Plds[wave][l16][kb + 8]);
    }
    // O += P @ V
#pragma unroll
    for (int nt = 0; nt < 8; ++nt) {
      v16h vf;
#pragma unroll
      for (int j = 0; j < 16; ++j) {
        int key = kp0 + kb2 + j;
        vf[j] = (_Float16)
            Vm[((size_t)(b * T + key) * H + h) * HD + nt * 16 + l16];
      }
      o_acc[nt] = wmma_f16(pf, vf, o_acc[nt]);
    }
  }
#pragma unroll
  for (int nt = 0; nt < 8; ++nt)
#pragma unroll
    for (int r = 0; r < 8; ++r) {
      int row = qbase + r + hi * 8;
      O[((size_t)(b * T + row) * H + h) * HD + nt * 16 + l16] =
          o_acc[nt][r] / lrow[r];
    }
}

// ---------------------------------------------------------------------------
// RMSNorm (optionally reading the previous token's row; t==0 -> zeros).
// ---------------------------------------------------------------------------
__global__ void rmsnorm_kernel(const float* __restrict__ X,
                               const float* __restrict__ w,
                               float* __restrict__ Y, int D, int T,
                               int shifted) {
  int row = blockIdx.x;
  int tid = threadIdx.x;
  const float* src;
  if (shifted) {
    int t = row % T;
    if (t == 0) {
      for (int j = tid; j < D; j += 256) Y[(size_t)row * D + j] = 0.0f;
      return;
    }
    src = X + (size_t)(row - 1) * D;
  } else {
    src = X + (size_t)row * D;
  }
  float s = 0.0f;
  for (int j = tid; j < D; j += 256) { float v = src[j]; s += v * v; }
  __shared__ float red[256];
  red[tid] = s;
  __syncthreads();
  for (int st = 128; st > 0; st >>= 1) {
    if (tid < st) red[tid] += red[tid + st];
    __syncthreads();
  }
  float rs = rsqrtf(red[0] / (float)D + EPS_C);
  for (int j = tid; j < D; j += 256)
    Y[(size_t)row * D + j] = src[j] * w[j] * rs;
}

// ---------------------------------------------------------------------------
// Qwen2 half-rotation RoPE, in place on [B,T,H,HD].
// ---------------------------------------------------------------------------
__global__ void rope_kernel(float* __restrict__ X, int total_pairs, int T,
                            int H) {
  int i = blockIdx.x * blockDim.x + threadIdx.x;
  if (i >= total_pairs) return;
  int d = i & 63;             // HD/2 = 64
  int h = (i >> 6) & (H - 1);
  int t = (i >> 10) % T;      // 64*H = 1024
  int b = i / (64 * H * T);
  float inv = __powf(10000.0f, -(2.0f * (float)d) / 128.0f);
  float ang = (float)t * inv;
  float c = __cosf(ang), s = __sinf(ang);
  float* p = X + ((size_t)(b * T + t) * H + h) * HD_C;
  float x0 = p[d], x1 = p[d + 64];
  p[d]      = x0 * c - x1 * s;
  p[d + 64] = x1 * c + x0 * s;
}

// ---------------------------------------------------------------------------
// Surprise gate: g = sigmoid(mean(ar^2) - mean((ar-pred)^2)), ar=proc-x.
// ---------------------------------------------------------------------------
__global__ void surprise_kernel(const float* __restrict__ proc,
                                const float* __restrict__ x,
                                const float* __restrict__ pred,
                                float* __restrict__ g, int D) {
  int row = blockIdx.x, tid = threadIdx.x;
  const float* pr = proc + (size_t)row * D;
  const float* xr = x + (size_t)row * D;
  const float* pd = pred + (size_t)row * D;
  float s1 = 0.0f, s2 = 0.0f;
  for (int j = tid; j < D; j += 256) {
    float ar = pr[j] - xr[j];
    float df = ar - pd[j];
    s1 += ar * ar;
    s2 += df * df;
  }
  __shared__ float a1[256], a2[256];
  a1[tid] = s1; a2[tid] = s2;
  __syncthreads();
  for (int st = 128; st > 0; st >>= 1) {
    if (tid < st) { a1[tid] += a1[tid + st]; a2[tid] += a2[tid + st]; }
    __syncthreads();
  }
  if (tid == 0) {
    float z = a1[0] / (float)D - a2[0] / (float)D;
    g[row] = 1.0f / (1.0f + __expf(-z));
  }
}

// ---------------------------------------------------------------------------
// Iterative top-K per batch row (K=256 of T=1024).
// ---------------------------------------------------------------------------
__global__ void topk_kernel(const float* __restrict__ g, int T, int Ksel,
                            int* __restrict__ idx_out,
                            float* __restrict__ val_out) {
  int b = blockIdx.x, tid = threadIdx.x;
  __shared__ float sv[1024];
  __shared__ float rv[256];
  __shared__ int ri[256];
  for (int j = tid; j < T; j += 256) sv[j] = g[(size_t)b * T + j];
  __syncthreads();
  for (int it = 0; it < Ksel; ++it) {
    float bv = -1e30f;
    int bi = 0;
    for (int j = tid; j < T; j += 256) {
      float v = sv[j];
      if (v > bv) { bv = v; bi = j; }
    }
    rv[tid] = bv; ri[tid] = bi;
    __syncthreads();
    for (int st = 128; st > 0; st >>= 1) {
      if (tid < st && rv[tid + st] > rv[tid]) {
        rv[tid] = rv[tid + st];
        ri[tid] = ri[tid + st];
      }
      __syncthreads();
    }
    if (tid == 0) {
      idx_out[b * Ksel + it] = ri[0];
      val_out[b * Ksel + it] = rv[0];
      sv[ri[0]] = -1e30f;
    }
    __syncthreads();
  }
}

__global__ void gather_kernel(const float* __restrict__ x,
                              const int* __restrict__ idx,
                              float* __restrict__ xsel, int D, int T,
                              int Ksel) {
  size_t j = (size_t)blockIdx.x * blockDim.x + threadIdx.x;
  int d = (int)(j % D);
  int row = (int)(j / D);
  int b = row / Ksel;
  int t = idx[row];
  xsel[j] = x[((size_t)b * T + t) * D + d];
}

__global__ void scatter_kernel(float* __restrict__ out,
                               const int* __restrict__ idx,
                               const float* __restrict__ gval,
                               const float* __restrict__ y, int D, int T,
                               int Ksel) {
  size_t j = (size_t)blockIdx.x * blockDim.x + threadIdx.x;
  int d = (int)(j % D);
  int row = (int)(j / D);
  int b = row / Ksel;
  int t = idx[row];
  out[((size_t)b * T + t) * D + d] += gval[row] * y[j];
}

__global__ void copy4_kernel(const float4* __restrict__ src,
                             float4* __restrict__ dst) {
  size_t i = (size_t)blockIdx.x * blockDim.x + threadIdx.x;
  dst[i] = src[i];
}

// ---------------------------------------------------------------------------
extern "C" void kernel_launch(void* const* d_in, const int* in_sizes, int n_in,
                              void* d_out, int out_size, void* d_ws,
                              size_t ws_size, hipStream_t stream) {
  (void)in_sizes; (void)n_in; (void)out_size; (void)ws_size;
  const float* x       = (const float*)d_in[0];
  const float* Wq      = (const float*)d_in[1];
  const float* bq      = (const float*)d_in[2];
  const float* Wk      = (const float*)d_in[3];
  const float* bk      = (const float*)d_in[4];
  const float* Wv      = (const float*)d_in[5];
  const float* bv      = (const float*)d_in[6];
  const float* Wo      = (const float*)d_in[7];
  const float* ln1_w   = (const float*)d_in[8];
  const float* ln2_w   = (const float*)d_in[9];
  const float* Wg      = (const float*)d_in[10];
  const float* Wu      = (const float*)d_in[11];
  const float* Wd      = (const float*)d_in[12];
  const float* tn_norm = (const float*)d_in[13];
  const float* tn_Wg   = (const float*)d_in[14];
  const float* tn_Wu   = (const float*)d_in[15];
  const float* tn_Wd   = (const float*)d_in[16];
  float* out = (float*)d_out;

  const int B = B_, T = T_C, D = D_C, H = H_C, F = F_C;
  const size_t BTD = (size_t)B * T * D;
  const size_t BTF = (size_t)B * T * F;
  const size_t BT  = (size_t)B * T;
  const size_t BK  = (size_t)B * KSEL;
  const size_t BKD = BK * D;

  float* w    = (float*)d_ws;
  float* fH   = w;                 // [B,T,D]
  float* fQ   = fH + BTD;          // [B,T,D]
  float* fK   = fQ + BTD;          // [B,T,D]
  float* fV   = fK + BTD;          // [B,T,D]
  float* fU   = fV + BTD;          // [B,T,F]
  float* fG   = fU + BTF;          // [B,T]
  float* fTV  = fG + BT;           // [B,K]
  float* fSel = fTV + BK;          // [B,K,D]
  float* fSeN = fSel + BKD;        // [B,K,D]
  float* fY   = fSeN + BKD;        // [B,K,D]
  int*   iIdx = (int*)(fY + BKD);  // [B,K]

  auto gemm = [&](const float* A, const float* Wm, const float* bias,
                  const float* aux, float* C, int M, int N, int Kd, int mode) {
    dim3 g(N / 128, M / 128);
    gemm_wmma_kernel<<<g, 256, 0, stream>>>(A, Wm, bias, aux, C, M, N, Kd,
                                            mode);
  };

  // ---- attention branch
  rmsnorm_kernel<<<BT, 256, 0, stream>>>(x, ln1_w, fH, D, T, 0);
  gemm(fH, Wq, bq, nullptr, fQ, B * T, D, D, 0);
  gemm(fH, Wk, bk, nullptr, fK, B * T, D, D, 0);
  gemm(fH, Wv, bv, nullptr, fV, B * T, D, D, 0);
  {
    int pairs = B * T * H * (HD_C / 2);
    rope_kernel<<<(pairs + 255) / 256, 256, 0, stream>>>(fQ, pairs, T, H);
    rope_kernel<<<(pairs + 255) / 256, 256, 0, stream>>>(fK, pairs, T, H);
  }
  {
    dim3 g(T / 64, B * H);
    flash_attn_kernel<<<g, 128, 0, stream>>>(fQ, fK, fV, fH, T, H);
  }
  gemm(fH, Wo, nullptr, x, fQ, B * T, D, D, 1);  // x1 = x + attn@Wo

  // ---- MLP branch: processed = x1 + swiglu(rmsnorm(x1))
  rmsnorm_kernel<<<BT, 256, 0, stream>>>(fQ, ln2_w, fK, D, T, 0);
  gemm(fK, Wu, nullptr, nullptr, fU, B * T, F, D, 0);   // u
  gemm(fK, Wg, nullptr, fU, fU, B * T, F, D, 2);        // act = silu(g)*u
  gemm(fU, Wd, nullptr, fQ, fV, B * T, D, F, 1);        // processed -> fV

  // ---- surprise: pred = tn_swiglu(rmsnorm(shift(processed)))
  rmsnorm_kernel<<<BT, 256, 0, stream>>>(fV, tn_norm, fH, D, T, 1);
  gemm(fH, tn_Wu, nullptr, nullptr, fU, B * T, F, D, 0);
  gemm(fH, tn_Wg, nullptr, fU, fU, B * T, F, D, 2);
  gemm(fU, tn_Wd, nullptr, nullptr, fK, B * T, D, F, 0);  // pred -> fK
  surprise_kernel<<<BT, 256, 0, stream>>>(fV, x, fK, fG, D);

  // ---- capacity routing: top-K, gather, process, scatter
  topk_kernel<<<B, 256, 0, stream>>>(fG, T, KSEL, iIdx, fTV);
  gather_kernel<<<(unsigned)(BKD / 256), 256, 0, stream>>>(x, iIdx, fSel, D, T,
                                                           KSEL);
  rmsnorm_kernel<<<BK, 256, 0, stream>>>(fSel, ln2_w, fSeN, D, T, 0);
  gemm(fSeN, Wu, nullptr, nullptr, fU, (int)BK, F, D, 0);
  gemm(fSeN, Wg, nullptr, fU, fU, (int)BK, F, D, 2);
  gemm(fU, Wd, nullptr, nullptr, fY, (int)BK, D, F, 0);

  copy4_kernel<<<(unsigned)(BTD / 4 / 256), 256, 0, stream>>>(
      (const float4*)x, (float4*)out);
  scatter_kernel<<<(unsigned)(BKD / 256), 256, 0, stream>>>(out, iIdx, fTV, fY,
                                                            D, T, KSEL);
}
